// AttentionHead_19902878449755
// MI455X (gfx1250) — matrix operand order repaired and checked by reading, not compiled, not run on hardware
//
#include <hip/hip_runtime.h>

// ---------------------------------------------------------------------------
// MI455X (gfx1250) single-head attention, wave32 + WMMA bf16 (f32 accum).
//  Pass 0: pre-split W into bf16 hi/lo, transposed [3][64][1024] (0.75 MB).
//  Pass 1: fused QKV projection, one wave = 16 rows x all 12 output tiles
//          (x split once, W frags = raw b128 loads). K row-major, V
//          transposed, both pre-split to bf16 hi/lo in workspace.
//  Pass 2: flash attention; 4 waves/WG share 32-key K/V blocks staged in
//          LDS with double-buffered GLOBAL_LOAD_ASYNC_TO_LDS_B128 (ASYNCcnt).
// Precision: hi/lo bf16 split => D += Ah*Bh + Ah*Bl + Al*Bh (~2^-15 rel).
// ---------------------------------------------------------------------------

typedef __attribute__((ext_vector_type(16))) __bf16        v16bf;
typedef __attribute__((ext_vector_type(8)))  unsigned int  v8u;
typedef __attribute__((ext_vector_type(8)))  float         v8f;

#define WMMA_BF16(A, B, C) \
  __builtin_amdgcn_wmma_f32_16x16x32_bf16(false, (A), false, (B), (short)0, (C), false, false)

constexpr int kT = 4096;   // sequence length
constexpr int kD = 64;     // head size
constexpr int kE = 1024;   // embedding dim
constexpr int kB = 4;      // batch

struct BF2 { unsigned short h, l; };

__device__ __forceinline__ BF2 split_bf(float f) {
  unsigned u = __builtin_bit_cast(unsigned, f);
  unsigned short h = (unsigned short)(u >> 16);            // truncate -> exact hi
  float r = f - __builtin_bit_cast(float, u & 0xFFFF0000u);
  unsigned short l = (unsigned short)(__builtin_bit_cast(unsigned, r) >> 16);
  return {h, l};
}

// Pack two floats' hi/lo bf16 into u32 pairs with v_perm_b32.
__device__ __forceinline__ void split_pair(float e0, float e1,
                                           unsigned& hi, unsigned& lo) {
  const unsigned u0 = __builtin_bit_cast(unsigned, e0);
  const unsigned u1 = __builtin_bit_cast(unsigned, e1);
  hi = __builtin_amdgcn_perm(u1, u0, 0x07060302u);         // {u1.hi16, u0.hi16}
  const float r0 = e0 - __builtin_bit_cast(float, u0 & 0xFFFF0000u);
  const float r1 = e1 - __builtin_bit_cast(float, u1 & 0xFFFF0000u);
  lo = __builtin_amdgcn_perm(__builtin_bit_cast(unsigned, r1),
                             __builtin_bit_cast(unsigned, r0), 0x07060302u);
}

// Async global->LDS copy, 16B per lane, tracked by ASYNCcnt (gfx1250).
// LDS dest = low 32 bits of the flat address of a __shared__ object
// (flat LDS aperture: addr[31:0] is the wave-relative LDS offset).
__device__ __forceinline__ void async_ld_b128(void* lds, const void* gsrc) {
  const unsigned laddr = (unsigned)(size_t)lds;
  asm volatile("global_load_async_to_lds_b128 %0, %1, off"
               :: "v"(laddr), "v"(gsrc)
               : "memory");
}

// ---------------------------------------------------------------------------
// Kernel 0: split W{q,k,v} into bf16 hi/lo, TRANSPOSED: wt[(m*64+n)*1024+k].
// ---------------------------------------------------------------------------
__global__ __launch_bounds__(256)
void wprep_kernel(const float* __restrict__ Wq, const float* __restrict__ Wk,
                  const float* __restrict__ Wv,
                  unsigned short* __restrict__ wt_hi,
                  unsigned short* __restrict__ wt_lo) {
  const int idx = blockIdx.x * 256 + threadIdx.x;          // [0, 3*64*1024)
  const int m   = idx / (kD * kE);
  const int rem = idx % (kD * kE);
  const int n   = rem / kE;
  const int k   = rem % kE;
  const float* W = (m == 0) ? Wq : (m == 1) ? Wk : Wv;
  const BF2 s = split_bf(W[(size_t)k * kD + n]);
  wt_hi[idx] = s.h;
  wt_lo[idx] = s.l;
}

// ---------------------------------------------------------------------------
// Kernel 1: fused QKV projection. One wave = one 16-row tile of x, all 12
// output tiles ({q,k,v} x four 16-col tiles). x split once per K-chunk.
//  A 16x32 bf16: row M=l16; VGPR j: K = half*16 + 2j, +1
//  B 32x16 bf16: col N=l16; VGPR j: K = half*16 + 2j, +1
//  C/D 16x16 f32: lane<16: N=l16, VGPR r->M=r ; lane>=16: M=8+r
// ---------------------------------------------------------------------------
__global__ __launch_bounds__(32)
void qkv_proj_kernel(const float* __restrict__ x,
                     const float* __restrict__ bq, const float* __restrict__ bk,
                     const float* __restrict__ bv,
                     const unsigned* __restrict__ wt_hi32,
                     const unsigned* __restrict__ wt_lo32,
                     float* __restrict__ qo,
                     unsigned short* __restrict__ khi, unsigned short* __restrict__ klo,
                     unsigned short* __restrict__ vhi, unsigned short* __restrict__ vlo) {
  const int lane = threadIdx.x & 31;
  const int half = lane >> 4;
  const int l16  = lane & 15;

  const int rowTile = blockIdx.x;
  const int arow    = rowTile * 16 + l16;

  v8f acc[12];
#pragma unroll
  for (int i = 0; i < 12; ++i) acc[i] = (v8f){0.f,0.f,0.f,0.f,0.f,0.f,0.f,0.f};

  for (int k0 = 0; k0 < kE; k0 += 32) {
    v8u ah, al;
#pragma unroll
    for (int j = 0; j < 8; ++j) {
      const int kk = k0 + half * 16 + 2 * j;
      const float2 xv = *(const float2*)(x + (size_t)arow * kE + kk);
      unsigned h, l;
      split_pair(xv.x, xv.y, h, l);
      ah[j] = h; al[j] = l;
    }
    const v16bf Ah = __builtin_bit_cast(v16bf, ah);
    const v16bf Al = __builtin_bit_cast(v16bf, al);

#pragma unroll
    for (int m = 0; m < 3; ++m) {
#pragma unroll
      for (int nt = 0; nt < 4; ++nt) {
        const int ncol = nt * 16 + l16;
        const size_t base = ((size_t)(m * kD + ncol)) * (kE / 2)
                          + (k0 >> 1) + half * 8;
        const v8u bh = *(const v8u*)(wt_hi32 + base);
        const v8u bl = *(const v8u*)(wt_lo32 + base);
        const v16bf Bh = __builtin_bit_cast(v16bf, bh);
        const v16bf Bl = __builtin_bit_cast(v16bf, bl);
        v8f a = acc[m * 4 + nt];
        a = WMMA_BF16(Ah, Bh, a);
        a = WMMA_BF16(Ah, Bl, a);
        a = WMMA_BF16(Al, Bh, a);
        acc[m * 4 + nt] = a;
      }
    }
  }

#pragma unroll
  for (int m = 0; m < 3; ++m) {
    const float* bias = (m == 0) ? bq : (m == 1) ? bk : bv;
#pragma unroll
    for (int nt = 0; nt < 4; ++nt) {
      const int col = nt * 16 + l16;
      const float bb = bias[col];
#pragma unroll
      for (int r = 0; r < 8; ++r) {
        const int row = rowTile * 16 + r + 8 * half;
        const float val = acc[m * 4 + nt][r] + bb;
        if (m == 0) {
          qo[(size_t)row * kD + col] = val;
        } else {
          const BF2 s = split_bf(val);
          if (m == 1) {
            khi[(size_t)row * kD + col] = s.h;
            klo[(size_t)row * kD + col] = s.l;
          } else {
            const int bidx = row >> 12;
            const int t    = row & (kT - 1);
            const size_t ti = ((size_t)bidx * kD + col) * kT + t;  // [B,64,T]
            vhi[ti] = s.h;
            vlo[ti] = s.l;
          }
        }
      }
    }
  }
}

// ---------------------------------------------------------------------------
// Kernel 2: flash attention. 128 threads = 4 waves; each wave owns one
// 16-query tile. 32-key K/V blocks are staged in LDS with double-buffered
// async global->LDS b128 copies (8 per thread per block -> wait asynccnt<=8).
//   kbuf : [32 keys][64 dims] bf16 (u32 = dim-pair)  -> score B-frag = v8u
//   vbuf : [64 dims][32 keys] bf16 (u32 = key-pair)  -> PV    B-frag = v8u
// ---------------------------------------------------------------------------
__global__ __launch_bounds__(128)
void attn_kernel(const float* __restrict__ qg,
                 const unsigned short* __restrict__ khi,
                 const unsigned short* __restrict__ klo,
                 const unsigned short* __restrict__ vhi,
                 const unsigned short* __restrict__ vlo,
                 float* __restrict__ out) {
  __shared__ uint4 kbh4[2][256], kbl4[2][256];   // 4 KB each buffer
  __shared__ uint4 vbh4[2][256], vbl4[2][256];
  __shared__ float pbuf[4 * 16 * 32];            // per-wave P staging

  const int tid  = threadIdx.x;
  const int wave = tid >> 5;
  const int lane = tid & 31;
  const int half = lane >> 4;
  const int l16  = lane & 15;

  const int b  = blockIdx.x / (kT / 64);
  const int qt = (blockIdx.x % (kT / 64)) * 4 + wave;

  const float*          qb  = qg  + (size_t)b * kT * kD;
  const unsigned short* khb = khi + (size_t)b * kT * kD;
  const unsigned short* klb = klo + (size_t)b * kT * kD;
  const unsigned short* vhb = vhi + (size_t)b * kD * kT;   // [64][T]
  const unsigned short* vlb = vlo + (size_t)b * kD * kT;

  float* pw = pbuf + wave * (16 * 32);

  // Issue the 8 async b128 copies (per thread) staging key-block `kt`.
  auto stage = [&](int kt, int buf) {
    const uint4* ksh = (const uint4*)khb + (size_t)kt * 8;  // 256 uint4
    const uint4* ksl = (const uint4*)klb + (size_t)kt * 8;
#pragma unroll
    for (int i = tid; i < 256; i += 128) {
      async_ld_b128(&kbh4[buf][i], ksh + i);
      async_ld_b128(&kbl4[buf][i], ksl + i);
    }
    const uint4* vsh = (const uint4*)vhb;  // dim-row stride: 512 uint4
    const uint4* vsl = (const uint4*)vlb;
#pragma unroll
    for (int i = tid; i < 256; i += 128) {
      const int dim = i >> 2, c4 = i & 3;
      async_ld_b128(&vbh4[buf][dim * 4 + c4], vsh + (size_t)dim * 512 + (kt >> 3) + c4);
      async_ld_b128(&vbl4[buf][dim * 4 + c4], vsl + (size_t)dim * 512 + (kt >> 3) + c4);
    }
  };

  // ---- q A-frags (two 16x32 chunks), scale folded, split once ----
  const float scale = 0.125f;  // 1/sqrt(64)
  v8u qh_u[2], ql_u[2];
  const int qrow = qt * 16 + l16;
#pragma unroll
  for (int c = 0; c < 2; ++c) {
#pragma unroll
    for (int j = 0; j < 8; ++j) {
      const int kk = c * 32 + half * 16 + 2 * j;
      const float2 qv = *(const float2*)(qb + (size_t)qrow * kD + kk);
      unsigned h, l;
      split_pair(qv.x * scale, qv.y * scale, h, l);
      qh_u[c][j] = h; ql_u[c][j] = l;
    }
  }

  v8f o[4];
#pragma unroll
  for (int nt = 0; nt < 4; ++nt) o[nt] = (v8f){0.f,0.f,0.f,0.f,0.f,0.f,0.f,0.f};
  float mrow[8], lrow[8];
#pragma unroll
  for (int r = 0; r < 8; ++r) { mrow[r] = -1e30f; lrow[r] = 0.f; }

  stage(0, 0);  // prologue: fill buffer 0

  for (int kt = 0, n = 0; kt < kT; kt += 32, ++n) {
    const int cur = n & 1;
    const bool more = (kt + 32 < kT);
    if (more) {
      stage(kt + 32, cur ^ 1);   // buf cur^1 last read in iter n-1 (barriered)
      asm volatile("s_wait_asynccnt 0x8" ::: "memory");  // block n landed
    } else {
      asm volatile("s_wait_asynccnt 0x0" ::: "memory");
    }
    __syncthreads();             // landed data visible to all 4 waves

    // ---- scores: two 16x16 C tiles over this 32-key block ----
    v8f s0 = {0.f,0.f,0.f,0.f,0.f,0.f,0.f,0.f};
    v8f s1 = {0.f,0.f,0.f,0.f,0.f,0.f,0.f,0.f};
#pragma unroll
    for (int c = 0; c < 2; ++c) {
      const int i0 = (l16)      * 4 + c * 2 + half;        // v8u units
      const int i1 = (16 + l16) * 4 + c * 2 + half;
      const v16bf K0h = __builtin_bit_cast(v16bf, ((const v8u*)kbh4[cur])[i0]);
      const v16bf K0l = __builtin_bit_cast(v16bf, ((const v8u*)kbl4[cur])[i0]);
      const v16bf K1h = __builtin_bit_cast(v16bf, ((const v8u*)kbh4[cur])[i1]);
      const v16bf K1l = __builtin_bit_cast(v16bf, ((const v8u*)kbl4[cur])[i1]);
      const v16bf Qh = __builtin_bit_cast(v16bf, qh_u[c]);
      const v16bf Ql = __builtin_bit_cast(v16bf, ql_u[c]);
      s0 = WMMA_BF16(Qh, K0h, s0);
      s0 = WMMA_BF16(Qh, K0l, s0);
      s0 = WMMA_BF16(Ql, K0h, s0);
      s1 = WMMA_BF16(Qh, K1h, s1);
      s1 = WMMA_BF16(Qh, K1l, s1);
      s1 = WMMA_BF16(Ql, K1h, s1);
    }

    // ---- online softmax (xor-shuffles stay inside each 16-lane half) ----
    float alpha[8];
#pragma unroll
    for (int r = 0; r < 8; ++r) {
      float t = fmaxf(s0[r], s1[r]);
      t = fmaxf(t, __shfl_xor(t, 1));
      t = fmaxf(t, __shfl_xor(t, 2));
      t = fmaxf(t, __shfl_xor(t, 4));
      t = fmaxf(t, __shfl_xor(t, 8));
      const float mn = fmaxf(mrow[r], t);
      alpha[r] = __expf(mrow[r] - mn);
      mrow[r]  = mn;
      const float p0 = __expf(s0[r] - mn);
      const float p1 = __expf(s1[r] - mn);
      pw[(r + 8 * half) * 32 + l16]      = p0;
      pw[(r + 8 * half) * 32 + 16 + l16] = p1;
      float rs = p0 + p1;
      rs += __shfl_xor(rs, 1);
      rs += __shfl_xor(rs, 2);
      rs += __shfl_xor(rs, 4);
      rs += __shfl_xor(rs, 8);
      lrow[r] = lrow[r] * alpha[r] + rs;
    }

    // ---- P as one 16x32 A-frag (from per-wave LDS), split via v_perm ----
    v8u ph_u, pl_u;
#pragma unroll
    for (int j = 0; j < 8; ++j) {
      const int kk = half * 16 + 2 * j;
      const float2 pv = *(const float2*)(&pw[l16 * 32 + kk]);
      unsigned h, l;
      split_pair(pv.x, pv.y, h, l);
      ph_u[j] = h; pl_u[j] = l;
    }
    const v16bf Ph = __builtin_bit_cast(v16bf, ph_u);
    const v16bf Pl = __builtin_bit_cast(v16bf, pl_u);

    // ---- O = O*alpha + P @ V ----
#pragma unroll
    for (int nt = 0; nt < 4; ++nt) {
#pragma unroll
      for (int r = 0; r < 8; ++r) o[nt][r] = o[nt][r] * alpha[r];
      const int iv = (nt * 16 + l16) * 2 + half;           // v8u units
      const v16bf Vh = __builtin_bit_cast(v16bf, ((const v8u*)vbh4[cur])[iv]);
      const v16bf Vl = __builtin_bit_cast(v16bf, ((const v8u*)vbl4[cur])[iv]);
      o[nt] = WMMA_BF16(Ph, Vh, o[nt]);
      o[nt] = WMMA_BF16(Ph, Vl, o[nt]);
      o[nt] = WMMA_BF16(Pl, Vh, o[nt]);
    }
    __syncthreads();  // all waves done with buffer `cur` (reused at n+2)
  }

  // ---- normalize and store ----
#pragma unroll
  for (int r = 0; r < 8; ++r) {
    const float inv  = 1.0f / lrow[r];
    const int   orow = qt * 16 + r + 8 * half;
#pragma unroll
    for (int nt = 0; nt < 4; ++nt) {
      out[(size_t)b * kT * kD + (size_t)orow * kD + nt * 16 + l16] =
          o[nt][r] * inv;
    }
  }
}

// ---------------------------------------------------------------------------
extern "C" void kernel_launch(void* const* d_in, const int* in_sizes, int n_in,
                              void* d_out, int out_size, void* d_ws,
                              size_t ws_size, hipStream_t stream) {
  (void)in_sizes; (void)n_in; (void)out_size; (void)ws_size;
  const float* x  = (const float*)d_in[0];
  const float* Wq = (const float*)d_in[1];
  const float* bq = (const float*)d_in[2];
  const float* Wk = (const float*)d_in[3];
  const float* bk = (const float*)d_in[4];
  const float* Wv = (const float*)d_in[5];
  const float* bv = (const float*)d_in[6];
  float* out = (float*)d_out;

  // workspace layout (~12.75 MB):
  //   q f32 [B*T,64]           @ 0      (4 MB)
  //   k_hi bf16 [B*T,64]       @ +4 MB  (2 MB)
  //   k_lo bf16 [B*T,64]       @ +6 MB  (2 MB)
  //   v_hi bf16 [B,64,T]       @ +8 MB  (2 MB)  (transposed)
  //   v_lo bf16 [B,64,T]       @ +10 MB (2 MB)
  //   wt_hi bf16 [3][64][1024] @ +12 MB (384 KB) (transposed)
  //   wt_lo bf16 [3][64][1024] @ +12.375 MB (384 KB)
  char* ws = (char*)d_ws;
  const size_t NTOK = (size_t)kB * kT * kD;
  float*          qw   = (float*)ws;
  unsigned short* khw  = (unsigned short*)(ws + NTOK * 4);
  unsigned short* klw  = (unsigned short*)(ws + NTOK * 4 + NTOK * 2);
  unsigned short* vhw  = (unsigned short*)(ws + NTOK * 4 + NTOK * 4);
  unsigned short* vlw  = (unsigned short*)(ws + NTOK * 4 + NTOK * 6);
  unsigned short* wthw = (unsigned short*)(ws + NTOK * 4 + NTOK * 8);
  unsigned short* wtlw = (unsigned short*)(ws + NTOK * 4 + NTOK * 8
                                              + (size_t)3 * kD * kE * 2);

  // 0) W pre-split/transpose
  dim3 g0((3 * kD * kE) / 256), b0(256);
  wprep_kernel<<<g0, b0, 0, stream>>>(Wq, Wk, Wv, wthw, wtlw);

  // 1) fused QKV projection: one wave per 16-row tile
  dim3 g1(kB * kT / 16), b1(32);
  qkv_proj_kernel<<<g1, b1, 0, stream>>>(x, bq, bk, bv,
                                         (const unsigned*)wthw,
                                         (const unsigned*)wtlw,
                                         qw, khw, klw, vhw, vlw);

  // 2) flash attention: 4 waves per block, one block per 64 query rows
  dim3 g2(kB * (kT / 64)), b2(128);
  attn_kernel<<<g2, b2, 0, stream>>>(qw, khw, klw, vhw, vlw, out);
}
